// TFSpatioTemporalFocalModulation_69140383531334
// MI455X (gfx1250) — matrix operand order repaired and checked by reading, not compiled, not run on hardware
//
#include <hip/hip_runtime.h>
#include <hip/hip_bf16.h>
#include <math.h>

// ---------------------------------------------------------------------------
// TFSpatioTemporalFocalModulation for MI455X (gfx1250, wave32, WMMA bf16)
//
// All dense GEMMs run on v_wmma_f32_16x16x32_bf16 (f32 accumulate).
// Weights are pre-packed into per-lane WMMA B-fragment order (bf16) so each
// fragment is one contiguous 32-byte load. A-fragments are gathered as
// float4 runs and converted f32->bf16 in registers.
// Depthwise convs / gelu / gating stay in f32 VALU, vectorized x4 channels.
// ---------------------------------------------------------------------------

typedef __bf16 bf16;
typedef __attribute__((ext_vector_type(16))) __bf16 v16bf;
typedef __attribute__((ext_vector_type(8)))  float  v8f;

#define DEV static __device__ __forceinline__

// Problem constants
#define NPIX  200704L            // B*T*H*W = 64*56*56
#define HW    3136L              // 56*56
#define CCH   96

DEV float gelu_f(float x) { return 0.5f * x * (1.0f + erff(x * 0.7071067811865475f)); }

// WMMA fragment layouts (wave32, 16x16x32 bf16):
//  A (16x32, MxK): lane -> m=lane&15, h=lane>>4; element e holds
//                  K = (e<8 ? 8*h + e : 16 + 8*h + (e-8))  (two 8-runs)
//  B (32x16, KxN): lane -> n=lane&15, h=lane>>4; element e holds K = (e<8 ? 8*h + e : 16 + 8*h + (e-8))  (same as A)
//  C/D (16x16 f32): acc[rr] is row M = rr + 8*h, col N = lane&15

// Convert 8 consecutive f32 into fragment elements [base, base+8)
DEV void cvt8(v16bf& a, int base, const float* __restrict__ p) {
  const float4 f0 = *(const float4*)p;
  const float4 f1 = *(const float4*)(p + 4);
  a[base + 0] = (bf16)f0.x; a[base + 1] = (bf16)f0.y;
  a[base + 2] = (bf16)f0.z; a[base + 3] = (bf16)f0.w;
  a[base + 4] = (bf16)f1.x; a[base + 5] = (bf16)f1.y;
  a[base + 6] = (bf16)f1.z; a[base + 7] = (bf16)f1.w;
}
DEV void zero8(v16bf& a, int base) {
#pragma unroll
  for (int i = 0; i < 8; ++i) a[base + i] = (bf16)0.0f;
}

// Load 3 A-chunks (K=96) from a row-major f32 matrix with stride 96.
DEV void load_afrags96(const float* __restrict__ arow, int h, v16bf A[3]) {
#pragma unroll
  for (int kc = 0; kc < 3; ++kc) {
    cvt8(A[kc], 0, arow + kc * 32 + 8 * h);
    cvt8(A[kc], 8, arow + kc * 32 + 16 + 8 * h);
  }
}

// Load one pre-packed B fragment (contiguous 32B per lane).
DEV v16bf load_bpacked(const bf16* __restrict__ P, int ct, int kcn, int kc, int lane) {
  return *(const v16bf*)(P + (((long)ct * kcn + kc) * 32 + lane) * 16);
}

DEV v8f wmma_bf16(v16bf a, v16bf b, v8f c) {
  return __builtin_amdgcn_wmma_f32_16x16x32_bf16(false, a, false, b, (short)0, c,
                                                 false, false);
}

// ---------------------------------------------------------------------------
// Prep: pack a row-major f32 weight (K x M) into B-fragment order, zero-padded.
// P[((ct*kcn + kc)*32 + lane)*16 + e] = W[(kc*32 + 16*(e>>3) + 8*(lane>>4) + (e&7))*M + ct*16+(lane&15)]
// ---------------------------------------------------------------------------
__global__ void k_pack_b(const float* __restrict__ W, bf16* __restrict__ P,
                         int K, int M, int CT) {
  const int kcn = K / 32;
  const int total = CT * kcn * 32 * 16;
  int idx = blockIdx.x * 256 + threadIdx.x;
  if (idx >= total) return;
  const int e = idx & 15;
  const int lane = (idx >> 4) & 31;
  const int rest = idx >> 9;
  const int kc = rest % kcn;
  const int ct = rest / kcn;
  const int k = kc * 32 + (e >> 3) * 16 + (lane >> 4) * 8 + (e & 7);
  const int col = ct * 16 + (lane & 15);
  P[idx] = (col < M) ? (bf16)W[(long)k * M + col] : (bf16)0.0f;
}

__global__ void k_zero(float* __restrict__ p, int n) {
  int i = blockIdx.x * 256 + threadIdx.x;
  if (i < n) p[i] = 0.0f;
}

// ---------------------------------------------------------------------------
// K1: fused input GEMM  xs = x@Wf+bf (195 cols) and xt = x@Wft+bft (99 cols)
// Column tiles map 1:1 to destinations (96 and 192 are multiples of 16), so
// the epilogue is branch-free per segment.
// ---------------------------------------------------------------------------
__global__ __launch_bounds__(256)
void k_in_gemm(const float* __restrict__ x,
               const bf16* __restrict__ wfp, const float* __restrict__ bfv,
               const bf16* __restrict__ wftp, const float* __restrict__ bftv,
               float* __restrict__ q, float* __restrict__ ctxs, float* __restrict__ gs,
               float* __restrict__ ctxt, float* __restrict__ gt) {
  const int lane = threadIdx.x & 31, wave = threadIdx.x >> 5;
  const int m = lane & 15, h = lane >> 4, n = lane & 15;
  const long r0 = ((long)blockIdx.x * 8 + wave) * 16;

  v16bf A[3];
  load_afrags96(x + (r0 + m) * CCH, h, A);

  // --- spatial: q columns (tiles 0..5) ---
#pragma unroll
  for (int ct = 0; ct < 6; ++ct) {
    v8f acc = {};
#pragma unroll
    for (int kc = 0; kc < 3; ++kc)
      acc = wmma_bf16(A[kc], load_bpacked(wfp, ct, 3, kc, lane), acc);
    const int col = ct * 16 + n;
    const float bias = bfv[col];
#pragma unroll
    for (int rr = 0; rr < 8; ++rr)
      q[(r0 + rr + 8 * h) * CCH + col] = acc[rr] + bias;
  }
  // --- spatial: ctx columns (tiles 6..11) ---
#pragma unroll
  for (int ct = 6; ct < 12; ++ct) {
    v8f acc = {};
#pragma unroll
    for (int kc = 0; kc < 3; ++kc)
      acc = wmma_bf16(A[kc], load_bpacked(wfp, ct, 3, kc, lane), acc);
    const int col = ct * 16 + n;
    const float bias = bfv[col];
#pragma unroll
    for (int rr = 0; rr < 8; ++rr)
      ctxs[(r0 + rr + 8 * h) * CCH + (col - 96)] = acc[rr] + bias;
  }
  // --- spatial: gate columns (tile 12, cols 192..194) ---
  {
    v8f acc = {};
#pragma unroll
    for (int kc = 0; kc < 3; ++kc)
      acc = wmma_bf16(A[kc], load_bpacked(wfp, 12, 3, kc, lane), acc);
    const int col = 192 + n;
    if (col < 195) {
      const float bias = bfv[col];
#pragma unroll
      for (int rr = 0; rr < 8; ++rr)
        gs[(r0 + rr + 8 * h) * 3 + (col - 192)] = acc[rr] + bias;
    }
  }
  // --- temporal: ctx columns (tiles 0..5) ---
#pragma unroll
  for (int ct = 0; ct < 6; ++ct) {
    v8f acc = {};
#pragma unroll
    for (int kc = 0; kc < 3; ++kc)
      acc = wmma_bf16(A[kc], load_bpacked(wftp, ct, 3, kc, lane), acc);
    const int col = ct * 16 + n;
    const float bias = bftv[col];
#pragma unroll
    for (int rr = 0; rr < 8; ++rr)
      ctxt[(r0 + rr + 8 * h) * CCH + col] = acc[rr] + bias;
  }
  // --- temporal: gate columns (tile 6, cols 96..98) ---
  {
    v8f acc = {};
#pragma unroll
    for (int kc = 0; kc < 3; ++kc)
      acc = wmma_bf16(A[kc], load_bpacked(wftp, 6, 3, kc, lane), acc);
    const int col = 96 + n;
    if (col < 99) {
      const float bias = bftv[col];
#pragma unroll
      for (int rr = 0; rr < 8; ++rr)
        gt[(r0 + rr + 8 * h) * 3 + (col - 96)] = acc[rr] + bias;
    }
  }
}

// ---------------------------------------------------------------------------
// K2: spatial depthwise 3x3 + gelu (f32 VALU, 4 channels per thread)
// ---------------------------------------------------------------------------
__global__ __launch_bounds__(256)
void k_dw3(const float* __restrict__ src, const float* __restrict__ kw,
           float* __restrict__ dst, long total4) {
  long idx = (long)blockIdx.x * 256 + threadIdx.x;
  if (idx >= total4) return;
  const int c = (int)(idx % 24) * 4;
  long p = idx / 24;
  const int w = (int)(p % 56); p /= 56;
  const int hh = (int)(p % 56);
  const long nimg = p / 56;
  float4 s = {0.0f, 0.0f, 0.0f, 0.0f};
#pragma unroll
  for (int dy = 0; dy < 3; ++dy) {
    const int y = hh + dy - 1;
    if ((unsigned)y >= 56u) continue;
#pragma unroll
    for (int dx = 0; dx < 3; ++dx) {
      const int xx = w + dx - 1;
      if ((unsigned)xx >= 56u) continue;
      const float4 v = *(const float4*)(src + ((nimg * 56 + y) * 56 + xx) * CCH + c);
      const float4 kk = *(const float4*)(kw + (dy * 3 + dx) * CCH + c);
      s.x += v.x * kk.x; s.y += v.y * kk.y; s.z += v.z * kk.z; s.w += v.w * kk.w;
    }
  }
  float4 o;
  o.x = gelu_f(s.x); o.y = gelu_f(s.y); o.z = gelu_f(s.z); o.w = gelu_f(s.w);
  *(float4*)(dst + idx * 4) = o;
}

// ---------------------------------------------------------------------------
// K3: spatial depthwise 5x5 + gelu + gate combine + per-image channel sums
// ---------------------------------------------------------------------------
__global__ __launch_bounds__(256)
void k_dw5(const float* __restrict__ ctx1, const float* __restrict__ kw,
           const float* __restrict__ gates, float* __restrict__ outAll,
           float* __restrict__ sum_s) {
  __shared__ float lsum[CCH];
  const int nimg = blockIdx.x / 56;
  const int hh = blockIdx.x % 56;
  for (int i = threadIdx.x; i < CCH; i += blockDim.x) lsum[i] = 0.0f;
  __syncthreads();
  for (int i = threadIdx.x; i < 56 * 24; i += blockDim.x) {
    const int c = (i % 24) * 4;
    const int w = i / 24;
    float4 s = {0.0f, 0.0f, 0.0f, 0.0f};
#pragma unroll
    for (int dy = 0; dy < 5; ++dy) {
      const int y = hh + dy - 2;
      if ((unsigned)y >= 56u) continue;
#pragma unroll
      for (int dx = 0; dx < 5; ++dx) {
        const int xx = w + dx - 2;
        if ((unsigned)xx >= 56u) continue;
        const float4 v =
            *(const float4*)(ctx1 + (((long)nimg * 56 + y) * 56 + xx) * CCH + c);
        const float4 kk = *(const float4*)(kw + (dy * 5 + dx) * CCH + c);
        s.x += v.x * kk.x; s.y += v.y * kk.y; s.z += v.z * kk.z; s.w += v.w * kk.w;
      }
    }
    float4 c2;
    c2.x = gelu_f(s.x); c2.y = gelu_f(s.y); c2.z = gelu_f(s.z); c2.w = gelu_f(s.w);
    const long row = ((long)nimg * 56 + hh) * 56 + w;
    const float4 c1 = *(const float4*)(ctx1 + row * CCH + c);
    const float g0 = gates[row * 3 + 0], g1 = gates[row * 3 + 1];
    float4 oa;
    oa.x = c1.x * g0 + c2.x * g1; oa.y = c1.y * g0 + c2.y * g1;
    oa.z = c1.z * g0 + c2.z * g1; oa.w = c1.w * g0 + c2.w * g1;
    *(float4*)(outAll + row * CCH + c) = oa;
    atomicAdd(&lsum[c + 0], c2.x);
    atomicAdd(&lsum[c + 1], c2.y);
    atomicAdd(&lsum[c + 2], c2.z);
    atomicAdd(&lsum[c + 3], c2.w);
  }
  __syncthreads();
  for (int i = threadIdx.x; i < CCH; i += blockDim.x)
    atomicAdd(&sum_s[(long)nimg * CCH + i], lsum[i]);
}

// ---------------------------------------------------------------------------
// K4: mod_s = (ctx_all_s + gelu(mean)*g2) @ Wh + bh   (global ctx fused in A)
// ---------------------------------------------------------------------------
__global__ __launch_bounds__(256)
void k_gemm_mods(const float* __restrict__ ctxAll, const float* __restrict__ sum_s,
                 const float* __restrict__ gates, const bf16* __restrict__ Whp,
                 const float* __restrict__ bh, float* __restrict__ mod) {
  const int lane = threadIdx.x & 31, wave = threadIdx.x >> 5;
  const int m = lane & 15, h = lane >> 4, n = lane & 15;
  const long r0 = ((long)blockIdx.x * 8 + wave) * 16;
  const long r = r0 + m;
  const long nimg = r / HW;
  const float g2 = gates[r * 3 + 2];
  const float invHW = 1.0f / 3136.0f;

  v16bf A[3];
#pragma unroll
  for (int kc = 0; kc < 3; ++kc) {
#pragma unroll
    for (int half = 0; half < 2; ++half) {
      const int kb = kc * 32 + half * 16 + 8 * h;
      const float4 f0 = *(const float4*)(ctxAll + r * CCH + kb);
      const float4 f1 = *(const float4*)(ctxAll + r * CCH + kb + 4);
      const float4 s0 = *(const float4*)(sum_s + nimg * CCH + kb);
      const float4 s1 = *(const float4*)(sum_s + nimg * CCH + kb + 4);
      const int b = half * 8;
      A[kc][b + 0] = (bf16)(f0.x + gelu_f(s0.x * invHW) * g2);
      A[kc][b + 1] = (bf16)(f0.y + gelu_f(s0.y * invHW) * g2);
      A[kc][b + 2] = (bf16)(f0.z + gelu_f(s0.z * invHW) * g2);
      A[kc][b + 3] = (bf16)(f0.w + gelu_f(s0.w * invHW) * g2);
      A[kc][b + 4] = (bf16)(f1.x + gelu_f(s1.x * invHW) * g2);
      A[kc][b + 5] = (bf16)(f1.y + gelu_f(s1.y * invHW) * g2);
      A[kc][b + 6] = (bf16)(f1.z + gelu_f(s1.z * invHW) * g2);
      A[kc][b + 7] = (bf16)(f1.w + gelu_f(s1.w * invHW) * g2);
    }
  }

#pragma unroll
  for (int ct = 0; ct < 6; ++ct) {
    v8f acc = {};
#pragma unroll
    for (int kc = 0; kc < 3; ++kc)
      acc = wmma_bf16(A[kc], load_bpacked(Whp, ct, 3, kc, lane), acc);
    const int col = ct * 16 + n;
    const float bias = bh[col];
#pragma unroll
    for (int rr = 0; rr < 8; ++rr)
      mod[(r0 + rr + 8 * h) * CCH + col] = acc[rr] + bias;
  }
}

// ---------------------------------------------------------------------------
// K5/K6: temporal conv1d over T=8 as WMMA GEMM with K = WTAP*96.
// im2col A-gather: shifting t by d moves the row by d*HW (zero outside [0,8)).
// 8-element runs never cross a 96 (dt) boundary, so float4 loads are safe.
// FUSE: second conv also writes ctx_all_t = ctx1*g0 + ctx2*g1.
// ---------------------------------------------------------------------------
template <int WTAP, bool FUSE>
__global__ __launch_bounds__(256)
void k_tconv(const float* __restrict__ src, const bf16* __restrict__ kwp,
             const float* __restrict__ gates, float* __restrict__ dst,
             float* __restrict__ outAll) {
  const int lane = threadIdx.x & 31, wave = threadIdx.x >> 5;
  const int m = lane & 15, h = lane >> 4, n = lane & 15;
  const long r0 = ((long)blockIdx.x * 8 + wave) * 16;
  const long r = r0 + m;
  const int t = (int)((r / HW) & 7);
  const int PAD = WTAP / 2;
  const int KC = WTAP * 3;  // K = WTAP*96 in chunks of 32

  v16bf A[KC];
#pragma unroll
  for (int kc = 0; kc < KC; ++kc) {
#pragma unroll
    for (int half = 0; half < 2; ++half) {
      const int kb = kc * 32 + half * 16 + 8 * h;
      const int dt = kb / CCH;
      const int c = kb - dt * CCH;
      const int dte = dt - PAD;
      if ((unsigned)(t + dte) < 8u)
        cvt8(A[kc], half * 8, src + (r + (long)dte * HW) * CCH + c);
      else
        zero8(A[kc], half * 8);
    }
  }

#pragma unroll
  for (int ct = 0; ct < 6; ++ct) {
    v8f acc = {};
#pragma unroll
    for (int kc = 0; kc < KC; ++kc)
      acc = wmma_bf16(A[kc], load_bpacked(kwp, ct, KC, kc, lane), acc);
    const int col = ct * 16 + n;
#pragma unroll
    for (int rr = 0; rr < 8; ++rr) {
      const long row = r0 + rr + 8 * h;
      const float ctx2 = gelu_f(acc[rr]);
      dst[row * CCH + col] = ctx2;
      if (FUSE) {
        const float g0 = gates[row * 3 + 0], g1 = gates[row * 3 + 1];
        const float c1 = src[row * CCH + col];
        outAll[row * CCH + col] = c1 * g0 + ctx2 * g1;
      }
    }
  }
}

// ---------------------------------------------------------------------------
// K7: temporal global context  ctx_all_t += gelu(mean_T(ctx2)) * g2
// ---------------------------------------------------------------------------
__global__ __launch_bounds__(256)
void k_tglob(const float* __restrict__ ctx2, const float* __restrict__ gates,
             float* __restrict__ outAll) {
  long idx = (long)blockIdx.x * 256 + threadIdx.x;
  if (idx >= 25088L * 24) return;
  const int c = (int)(idx % 24) * 4;
  const long site = idx / 24;
  const long b = site / HW;
  const long hw = site % HW;
  float4 s = {0.0f, 0.0f, 0.0f, 0.0f};
#pragma unroll
  for (int t = 0; t < 8; ++t) {
    const float4 v = *(const float4*)(ctx2 + ((b * 8 + t) * HW + hw) * CCH + c);
    s.x += v.x; s.y += v.y; s.z += v.z; s.w += v.w;
  }
  float4 mg;
  mg.x = gelu_f(s.x * 0.125f); mg.y = gelu_f(s.y * 0.125f);
  mg.z = gelu_f(s.z * 0.125f); mg.w = gelu_f(s.w * 0.125f);
#pragma unroll
  for (int t = 0; t < 8; ++t) {
    const long row = (b * 8 + t) * HW + hw;
    const float g = gates[row * 3 + 2];
    float4 o = *(float4*)(outAll + row * CCH + c);
    o.x += mg.x * g; o.y += mg.y * g; o.z += mg.z * g; o.w += mg.w * g;
    *(float4*)(outAll + row * CCH + c) = o;
  }
}

// ---------------------------------------------------------------------------
// K8: plain GEMM  out = A @ W(96x96 packed bf16) + bias
// ---------------------------------------------------------------------------
__global__ __launch_bounds__(256)
void k_gemm_plain(const float* __restrict__ Ain, const bf16* __restrict__ Wp,
                  const float* __restrict__ bias, float* __restrict__ Cout) {
  const int lane = threadIdx.x & 31, wave = threadIdx.x >> 5;
  const int m = lane & 15, h = lane >> 4, n = lane & 15;
  const long r0 = ((long)blockIdx.x * 8 + wave) * 16;

  v16bf A[3];
  load_afrags96(Ain + (r0 + m) * CCH, h, A);

#pragma unroll
  for (int ct = 0; ct < 6; ++ct) {
    v8f acc = {};
#pragma unroll
    for (int kc = 0; kc < 3; ++kc)
      acc = wmma_bf16(A[kc], load_bpacked(Wp, ct, 3, kc, lane), acc);
    const int col = ct * 16 + n;
    const float b = bias[col];
#pragma unroll
    for (int rr = 0; rr < 8; ++rr)
      Cout[(r0 + rr + 8 * h) * CCH + col] = acc[rr] + b;
  }
}

// ---------------------------------------------------------------------------
// K9: out = (q * mod_s * mod_t) @ Wproj + bproj  (elementwise fused in A)
// ---------------------------------------------------------------------------
__global__ __launch_bounds__(256)
void k_final(const float* __restrict__ q, const float* __restrict__ ms,
             const float* __restrict__ mt, const bf16* __restrict__ Wpp,
             const float* __restrict__ bp, float* __restrict__ out) {
  const int lane = threadIdx.x & 31, wave = threadIdx.x >> 5;
  const int m = lane & 15, h = lane >> 4, n = lane & 15;
  const long r0 = ((long)blockIdx.x * 8 + wave) * 16;
  const long r = r0 + m;

  v16bf A[3];
#pragma unroll
  for (int kc = 0; kc < 3; ++kc) {
#pragma unroll
    for (int half = 0; half < 2; ++half) {
      const int kb = kc * 32 + half * 16 + 8 * h;
      const int b = half * 8;
#pragma unroll
      for (int part = 0; part < 2; ++part) {
        const float4 fq = *(const float4*)(q + r * CCH + kb + part * 4);
        const float4 fs = *(const float4*)(ms + r * CCH + kb + part * 4);
        const float4 ft = *(const float4*)(mt + r * CCH + kb + part * 4);
        A[kc][b + part * 4 + 0] = (bf16)(fq.x * fs.x * ft.x);
        A[kc][b + part * 4 + 1] = (bf16)(fq.y * fs.y * ft.y);
        A[kc][b + part * 4 + 2] = (bf16)(fq.z * fs.z * ft.z);
        A[kc][b + part * 4 + 3] = (bf16)(fq.w * fs.w * ft.w);
      }
    }
  }

#pragma unroll
  for (int ct = 0; ct < 6; ++ct) {
    v8f acc = {};
#pragma unroll
    for (int kc = 0; kc < 3; ++kc)
      acc = wmma_bf16(A[kc], load_bpacked(Wpp, ct, 3, kc, lane), acc);
    const int col = ct * 16 + n;
    const float b = bp[col];
#pragma unroll
    for (int rr = 0; rr < 8; ++rr)
      out[(r0 + rr + 8 * h) * CCH + col] = acc[rr] + b;
  }
}

// ---------------------------------------------------------------------------
// Host launcher
// ---------------------------------------------------------------------------
extern "C" void kernel_launch(void* const* d_in, const int* in_sizes, int n_in,
                              void* d_out, int out_size, void* d_ws, size_t ws_size,
                              hipStream_t stream) {
  (void)in_sizes; (void)n_in; (void)out_size; (void)ws_size;
  const float* x    = (const float*)d_in[0];
  const float* Wf   = (const float*)d_in[1];
  const float* bfv  = (const float*)d_in[2];
  const float* Wft  = (const float*)d_in[3];
  const float* bftv = (const float*)d_in[4];
  const float* k2d0 = (const float*)d_in[5];
  const float* k2d1 = (const float*)d_in[6];
  const float* k1d0 = (const float*)d_in[7];
  const float* k1d1 = (const float*)d_in[8];
  const float* Wh   = (const float*)d_in[9];
  const float* bh   = (const float*)d_in[10];
  const float* Wht  = (const float*)d_in[11];
  const float* bht  = (const float*)d_in[12];
  const float* Wp   = (const float*)d_in[13];
  const float* bp   = (const float*)d_in[14];
  float* out = (float*)d_out;

  const long NC = NPIX * CCH;
  char* ws = (char*)d_ws;
  size_t off = 0;
  auto take = [&](size_t bytes) -> char* {
    char* p = ws + off;
    off = (off + bytes + 255) & ~(size_t)255;
    return p;
  };
  // Big N x 96 f32 slots, reused by liveness:
  float* S0 = (float*)take(NC * 4);  // q
  float* S1 = (float*)take(NC * 4);  // ctx_s   -> mod_s
  float* S2 = (float*)take(NC * 4);  // ctx_t0  -> ctx2_t
  float* S3 = (float*)take(NC * 4);  // ctx1_s  -> ctx1_t -> mod_t
  float* S4 = (float*)take(NC * 4);  // ctx_all_s -> ctx_all_t
  float* gs = (float*)take(NPIX * 3 * 4);
  float* gt = (float*)take(NPIX * 3 * 4);
  float* sum_s = (float*)take(64 * CCH * 4);
  // Packed bf16 B-fragment weights: CT * (K/32) * 512 elements each.
  bf16* wfp  = (bf16*)take(13 * 3 * 512 * 2);   // 96 x 195
  bf16* wftp = (bf16*)take(7 * 3 * 512 * 2);    // 96 x 99
  bf16* k1p  = (bf16*)take(6 * 9 * 512 * 2);    // 288 x 96
  bf16* k2p  = (bf16*)take(6 * 15 * 512 * 2);   // 480 x 96
  bf16* whp  = (bf16*)take(6 * 3 * 512 * 2);    // 96 x 96
  bf16* whtp = (bf16*)take(6 * 3 * 512 * 2);
  bf16* wpp  = (bf16*)take(6 * 3 * 512 * 2);

  auto pack = [&](const float* s, bf16* d, int K, int M, int CT) {
    const int total = CT * (K / 32) * 512;
    k_pack_b<<<(total + 255) / 256, 256, 0, stream>>>(s, d, K, M, CT);
  };
  pack(Wf, wfp, 96, 195, 13);
  pack(Wft, wftp, 96, 99, 7);
  pack(k1d0, k1p, 288, 96, 6);
  pack(k1d1, k2p, 480, 96, 6);
  pack(Wh, whp, 96, 96, 6);
  pack(Wht, whtp, 96, 96, 6);
  pack(Wp, wpp, 96, 96, 6);
  k_zero<<<(64 * CCH + 255) / 256, 256, 0, stream>>>(sum_s, 64 * CCH);

  const int GEMM_GRID = (int)(NPIX / 16 / 8);  // 1568 blocks of 8 waves

  // K1: fused input GEMMs
  k_in_gemm<<<GEMM_GRID, 256, 0, stream>>>(x, wfp, bfv, wftp, bftv, S0, S1, gs, S2, gt);
  // K2: spatial dw3x3 + gelu
  k_dw3<<<(int)((NC / 4 + 255) / 256), 256, 0, stream>>>(S1, k2d0, S3, NC / 4);
  // K3: spatial dw5x5 + gelu + gate combine + channel sums
  k_dw5<<<64 * 56, 256, 0, stream>>>(S3, k2d1, gs, S4, sum_s);
  // K4: mod_s GEMM with fused global context
  k_gemm_mods<<<GEMM_GRID, 256, 0, stream>>>(S4, sum_s, gs, whp, bh, S1);
  // K5: temporal conv1d width 3 (K=288 GEMM) -> ctx1_t
  k_tconv<3, false><<<GEMM_GRID, 256, 0, stream>>>(S2, k1p, nullptr, S3, nullptr);
  // K6: temporal conv1d width 5 (K=480 GEMM) + gate combine -> ctx2_t, ctx_all_t
  k_tconv<5, true><<<GEMM_GRID, 256, 0, stream>>>(S3, k2p, gt, S2, S4);
  // K7: temporal global context
  k_tglob<<<(int)((25088L * 24 + 255) / 256), 256, 0, stream>>>(S2, gt, S4);
  // K8: mod_t GEMM
  k_gemm_plain<<<GEMM_GRID, 256, 0, stream>>>(S4, whtp, bht, S3);
  // K9: final fused GEMM into d_out
  k_final<<<GEMM_GRID, 256, 0, stream>>>(S0, S1, S3, wpp, bp, out);
}
